// SimpleSNN_78632261255763
// MI455X (gfx1250) — matrix operand order repaired and checked by reading, not compiled, not run on hardware
//
#include <hip/hip_runtime.h>

typedef __bf16 bf16_t;
typedef __attribute__((ext_vector_type(16))) __bf16 v16bf;
typedef __attribute__((ext_vector_type(8)))  __bf16 v8bf;
typedef __attribute__((ext_vector_type(4)))  __bf16 v4bf;
typedef __attribute__((ext_vector_type(8)))  float  v8f;
typedef __attribute__((ext_vector_type(4)))  float  v4f;

#define SNN_TAU    0.9f
#define SNN_THRESH 1.0f
#define BM   64      // batch rows per block
#define HID  256     // hidden neurons
#define NIN  784     // input features
#define NOUT 10      // output classes
#define PAD  8       // bf16 row padding: keeps 16B alignment, skews banks
#define NCHUNK ((NIN + 31) / 32)   // 25 k-chunks (last one partial: 16 cols)

__device__ __forceinline__ v16bf frag_cat(v8bf lo, v8bf hi) {
    return __builtin_shufflevector(lo, hi, 0, 1, 2, 3, 4, 5, 6, 7,
                                           8, 9, 10, 11, 12, 13, 14, 15);
}

__device__ __forceinline__ v4bf cvt4(v4f f) {
    v4bf r;
    r[0] = (bf16_t)f[0]; r[1] = (bf16_t)f[1];
    r[2] = (bf16_t)f[2]; r[3] = (bf16_t)f[3];
    return r;
}

__global__ __launch_bounds__(256, 1)
void snn_fused_kernel(const float* __restrict__ x,
                      const float* __restrict__ W1,
                      const float* __restrict__ b1,
                      const float* __restrict__ W2,
                      const float* __restrict__ b2,
                      const int*   __restrict__ nsteps,
                      float* __restrict__ out,
                      int Btot)
{
    __shared__ bf16_t sW1[HID][32 + PAD];     // W1 k-chunk, natural row-major
    __shared__ bf16_t sX [BM][32 + PAD];      // x k-chunk, natural row-major
    __shared__ bf16_t sS1[BM][HID + PAD];     // spikes, [batch][hidden]
    __shared__ bf16_t sW2[16][HID + PAD];     // W2 natural [class][k], rows>=10 zero

    const int tid  = threadIdx.x;
    const int lane = tid & 31;
    const int wave = tid >> 5;
    const int row0 = blockIdx.x * BM;

    const int T = *nsteps;

    // ---- stage W2 once: [class][k], bf16, zero-pad classes 10..15 ----
    for (int q = tid; q < 16 * HID / 4; q += 256) {
        int o = (q * 4) / HID;
        int k = (q * 4) % HID;
        v4bf v = {};
        if (o < NOUT) v = cvt4(*(const v4f*)&W2[o * HID + k]);
        *(v4bf*)&sW2[o][k] = v;
    }

    // ---- per-wave tile assignment: D1 = W1 @ x^T is (hidden 256) x (batch 64)
    const int bt   = wave & 3;           // batch-tile (16 cols of D1)
    const int htb  = (wave >> 2) * 8;    // first of 8 hidden-tiles (16 rows each)
    const int mrow = lane & 15;          // M within tile for A fragments
    const int ncol = lane & 15;          // N within tile for B/C fragments
    const int kbA  = (lane < 16) ? 0 : 8;    // ISA-layout K base, unused (A uses kbB)
    const int kbB  = (lane < 16) ? 0 : 16;   // A/B-fragment K base (contiguous k = 16h + s)
    const int rhi  = (lane < 16) ? 0 : 8;    // C/D-fragment M offset

    // per-thread staging decomposition (fixed across chunks)
    int wn[8], wc[8];                    // W1: 8 quads/thread
#pragma unroll
    for (int i = 0; i < 8; ++i) {
        int q = tid + i * 256;           // 0..2047 over 256x32
        wn[i] = q >> 3;
        wc[i] = (q & 7) * 4;
    }
    int xr[2], xc[2];                    // x: 2 quads/thread
#pragma unroll
    for (int i = 0; i < 2; ++i) {
        int q = tid + i * 256;           // 0..511 over 64x32
        xr[i] = q >> 3;
        xc[i] = (q & 7) * 4;
    }

    v4bf nW1[8], nX[2];                  // register-staged next chunk (bf16)

    auto load_chunk = [&](int kk, bool tail) {
#pragma unroll
        for (int i = 0; i < 8; ++i) {
            int k = kk + wc[i];
            nW1[i] = (v4bf){};
            if (!tail || k < NIN)
                nW1[i] = cvt4(*(const v4f*)&W1[wn[i] * NIN + k]);
        }
#pragma unroll
        for (int i = 0; i < 2; ++i) {
            int k  = kk + xc[i];
            int rr = row0 + xr[i];
            nX[i] = (v4bf){};
            if ((!tail || k < NIN) && rr < Btot) {
                nX[i] = cvt4(*(const v4f*)&x[(size_t)rr * NIN + k]);
                if (kk + 32 < NIN)
                    __builtin_prefetch(&x[(size_t)rr * NIN + k + 32], 0, 1);
            }
        }
    };
    auto store_chunk = [&]() {
#pragma unroll
        for (int i = 0; i < 8; ++i) *(v4bf*)&sW1[wn[i]][wc[i]] = nW1[i];
#pragma unroll
        for (int i = 0; i < 2; ++i) *(v4bf*)&sX[xr[i]][xc[i]]  = nX[i];
    };

    v8f cur1[8] = {};

    // ---- layer-1 GEMM, software-pipelined:
    //      global->regs for chunk i+1 overlaps WMMAs on chunk i
    load_chunk(0, false);
    store_chunk();
    for (int ck = 0; ck < NCHUNK; ++ck) {
        __syncthreads();                         // chunk ck visible in LDS
        const bool more = (ck + 1) < NCHUNK;
        if (more) {
            int kkn = (ck + 1) * 32;
            load_chunk(kkn, kkn + 32 > NIN);     // only last chunk is guarded
        }

        // B fragment (x^T): fixed batch row, contiguous k
        const bf16_t* xrow = &sX[bt * 16 + ncol][0];
        v16bf bfr = frag_cat(*(const v8bf*)&xrow[kbB],
                             *(const v8bf*)&xrow[kbB + 8]);
        // all 8 A fragments first, then 8 back-to-back WMMAs
        v16bf afr[8];
#pragma unroll
        for (int t = 0; t < 8; ++t) {
            const bf16_t* wrow = &sW1[(htb + t) * 16 + mrow][0];
            afr[t] = frag_cat(*(const v8bf*)&wrow[kbB],
                              *(const v8bf*)&wrow[kbB + 8]);
        }
#pragma unroll
        for (int t = 0; t < 8; ++t)
            cur1[t] = __builtin_amdgcn_wmma_f32_16x16x32_bf16(
                false, afr[t], false, bfr, (short)0, cur1[t], false, false);

        __syncthreads();                         // all readers done with LDS
        if (more) store_chunk();
    }

    // add bias b1 (D1 rows = hidden -> 8 consecutive b1 values per lane)
#pragma unroll
    for (int t = 0; t < 8; ++t) {
        int hb = (htb + t) * 16 + rhi;
        v4f blo = *(const v4f*)&b1[hb];
        v4f bhi = *(const v4f*)&b1[hb + 4];
#pragma unroll
        for (int r = 0; r < 4; ++r) {
            cur1[t][r]     += blo[r];
            cur1[t][r + 4] += bhi[r];
        }
    }

    // ---- SNN time loop: cur1/m1 stay in VGPRs, spikes via b128 LDS ops ----
    v8f m1[8] = {};
    v8f m2  = {};
    v8f cnt = {};
    const float bias2 = (ncol < NOUT) ? b2[ncol] : 0.0f;

    for (int step = 0; step < T; ++step) {
        __syncthreads();   // sS1 readers from previous step are done
#pragma unroll
        for (int t = 0; t < 8; ++t) {
            v8bf sp;
#pragma unroll
            for (int r = 0; r < 8; ++r) {
                float m = SNN_TAU * m1[t][r] + cur1[t][r];
                float s = (m >= SNN_THRESH) ? 1.0f : 0.0f;
                m1[t][r] = m - s * SNN_THRESH;
                sp[r] = (bf16_t)s;
            }
            // lane holds 8 consecutive hidden values of one batch column:
            // one 16B store into sS1[batch][hidden]
            *(v8bf*)&sS1[bt * 16 + ncol][(htb + t) * 16 + rhi] = sp;
        }
        __syncthreads();

        if (wave < 4) {   // wave w owns batch-tile w of cur2 (16 rows x 16 classes)
            v8f c2 = {};
#pragma unroll
            for (int kc = 0; kc < 8; ++kc) {
                const bf16_t* srow = &sS1[wave * 16 + mrow][kc * 32];
                v16bf a2 = frag_cat(*(const v8bf*)&srow[kbB],
                                    *(const v8bf*)&srow[kbB + 8]);
                const bf16_t* wrow = &sW2[ncol][kc * 32];
                v16bf bw = frag_cat(*(const v8bf*)&wrow[kbB],
                                    *(const v8bf*)&wrow[kbB + 8]);
                c2 = __builtin_amdgcn_wmma_f32_16x16x32_bf16(
                    false, a2, false, bw, (short)0, c2, false, false);
            }
#pragma unroll
            for (int r = 0; r < 8; ++r) {
                float m = SNN_TAU * m2[r] + (c2[r] + bias2);
                float s = (m >= SNN_THRESH) ? 1.0f : 0.0f;
                m2[r] = m - s * SNN_THRESH;
                cnt[r] += s;
            }
        }
    }

    // ---- write spike counts (cur2 C-layout: M=batch row, N=class) ----
    if (wave < 4 && ncol < NOUT) {
#pragma unroll
        for (int r = 0; r < 8; ++r) {
            int row = row0 + wave * 16 + r + rhi;
            if (row < Btot)
                out[(size_t)row * NOUT + ncol] = cnt[r];
        }
    }
}

extern "C" void kernel_launch(void* const* d_in, const int* in_sizes, int n_in,
                              void* d_out, int out_size, void* d_ws, size_t ws_size,
                              hipStream_t stream) {
    const float* x  = (const float*)d_in[0];
    const float* W1 = (const float*)d_in[1];
    const float* b1 = (const float*)d_in[2];
    const float* W2 = (const float*)d_in[3];
    const float* b2 = (const float*)d_in[4];
    const int* nst  = (const int*)d_in[5];
    float* out = (float*)d_out;

    int Btot = in_sizes[0] / NIN;            // 32768
    int grid = (Btot + BM - 1) / BM;         // 512 blocks
    snn_fused_kernel<<<grid, 256, 0, stream>>>(x, W1, b1, W2, b2, nst, out, Btot);
}
